// S6_SSM_31490700214765
// MI455X (gfx1250) — compile-verified, matrix operand order checked by reading, not run on hardware
//
#include <hip/hip_runtime.h>
#include <hip/hip_bf16.h>
#include <math.h>

// ---------------- model dims ----------------
#define DM   256
#define DI   512
#define NS   16      // SSM state dim N
#define DTRK 16      // dt rank
#define NL   4
#define DOUT 10
#define BB   32      // batch
#define LL   196     // sequence length (14*14)
#define MM   (BB*LL) // 6272 rows, divisible by 32

typedef __attribute__((ext_vector_type(16))) __bf16 v16bf;
typedef __attribute__((ext_vector_type(8)))  __bf16 v8bf;
typedef __attribute__((ext_vector_type(8)))  float  v8f;

__device__ __forceinline__ float silu_f(float x)     { return x / (1.f + __expf(-x)); }
__device__ __forceinline__ float softplus_f(float x) { return x > 20.f ? x : log1pf(__expf(x)); }

// f32 -> bf16 bulk convert (weights, once per launch)
__global__ void cvt_bf16_kernel(const float* __restrict__ src, __bf16* __restrict__ dst, int n) {
    int i = blockIdx.x * 256 + threadIdx.x;
    if (i < n) dst[i] = (__bf16)src[i];
}

// ---------------------------------------------------------------------------
// WMMA GEMM, bf16 operands in memory, f32 accumulate:
//   C[M x N] = A[M x K] * W^T     (W is [N x K] row-major)
// Register blocking: one wave computes MT x NT 16x16 tiles (MT*NT WMMAs per
// K-step of 32, A/B fragments reused). Fragments loaded as 2x b128 per tile.
// Fragment layouts per CDNA5 ISA 7.12.2 (16-bit A 16x32 / mirrored B 32x16):
//   lanes 0-15: row/col = lane,    K = {k0..k0+7, k0+16..k0+23}
//   lanes16-31: row/col = lane-16, K = {k0+8..k0+15, k0+24..k0+31}
// C/D: lane&15 = N, VGPR v -> M = v + 8*(lane>=16)
// ---------------------------------------------------------------------------
template <int MT, int NT>
__global__ __launch_bounds__(32)
void gemm_bf16_wmma(const __bf16* __restrict__ A, const __bf16* __restrict__ W,
                    float* __restrict__ C, int M, int N, int K) {
    const int lane = threadIdx.x & 31;
    const int half = lane >> 4;
    const int lid  = lane & 15;
    const int koff = half * 8;

    const __bf16* arow[MT];
    const __bf16* wrow[NT];
#pragma unroll
    for (int mt = 0; mt < MT; ++mt)
        arow[mt] = A + (size_t)((blockIdx.x * MT + mt) * 16 + lid) * K;
#pragma unroll
    for (int nt = 0; nt < NT; ++nt)
        wrow[nt] = W + (size_t)((blockIdx.y * NT + nt) * 16 + lid) * K;

    const v8f zero = {};
    v8f acc[MT][NT];
#pragma unroll
    for (int mt = 0; mt < MT; ++mt)
#pragma unroll
        for (int nt = 0; nt < NT; ++nt) acc[mt][nt] = zero;

    for (int k0 = 0; k0 < K; k0 += 32) {
        v16bf a[MT], b[NT];
#pragma unroll
        for (int mt = 0; mt < MT; ++mt) {
            v8bf lo = *(const v8bf*)(arow[mt] + k0 + koff);
            v8bf hi = *(const v8bf*)(arow[mt] + k0 + 16 + koff);
            a[mt] = __builtin_shufflevector(lo, hi, 0,1,2,3,4,5,6,7,8,9,10,11,12,13,14,15);
        }
#pragma unroll
        for (int nt = 0; nt < NT; ++nt) {
            v8bf lo = *(const v8bf*)(wrow[nt] + k0 + koff);
            v8bf hi = *(const v8bf*)(wrow[nt] + k0 + 16 + koff);
            b[nt] = __builtin_shufflevector(lo, hi, 0,1,2,3,4,5,6,7,8,9,10,11,12,13,14,15);
        }
#pragma unroll
        for (int mt = 0; mt < MT; ++mt)
#pragma unroll
            for (int nt = 0; nt < NT; ++nt)
                acc[mt][nt] = __builtin_amdgcn_wmma_f32_16x16x32_bf16(
                    false, a[mt], false, b[nt], (short)0, acc[mt][nt], false, false);
    }

#pragma unroll
    for (int mt = 0; mt < MT; ++mt) {
#pragma unroll
        for (int nt = 0; nt < NT; ++nt) {
            const int n = (blockIdx.y * NT + nt) * 16 + lid;
#pragma unroll
            for (int v = 0; v < 8; ++v) {
                int m = (blockIdx.x * MT + mt) * 16 + v + 8 * half;
                C[(size_t)m * N + n] = acc[mt][nt][v];
            }
        }
    }
}

// ---------------------------------------------------------------------------
// Patch embedding: 16x16/stride16 conv as per-patch dot products.
// Output stored channel-major (B, DM, 196): raw-reshape-aliased to (B,196,DM).
// One block handles 4 patches (reuses pe_w reads 4x; pe_w lives in L2).
// ---------------------------------------------------------------------------
__global__ __launch_bounds__(256)
void patch_embed_kernel(const float* __restrict__ x, const float* __restrict__ pe_w,
                        const float* __restrict__ pe_b, float* __restrict__ conv_out) {
    __shared__ float sp[4 * 768];
    const int b  = blockIdx.x / 49;
    const int pg = blockIdx.x % 49;
    for (int idx = threadIdx.x; idx < 4 * 768; idx += 256) {
        int q = idx / 768, r = idx - q * 768;
        int ch = r >> 8, i = (r >> 4) & 15, j = r & 15;
        int p = pg * 4 + q, ph = p / 14, pw = p % 14;
        sp[idx] = x[(((size_t)b * 3 + ch) * 224 + ph * 16 + i) * 224 + pw * 16 + j];
    }
    __syncthreads();
    const int c = threadIdx.x;
    const float* __restrict__ wrow = pe_w + c * 768;
    float s0 = pe_b[c], s1 = s0, s2 = s0, s3 = s0;
    for (int k = 0; k < 768; ++k) {
        float w = wrow[k];
        s0 += w * sp[k];
        s1 += w * sp[768 + k];
        s2 += w * sp[1536 + k];
        s3 += w * sp[2304 + k];
    }
    size_t base = ((size_t)b * DM + c) * LL + pg * 4;
    conv_out[base + 0] = s0; conv_out[base + 1] = s1;
    conv_out[base + 2] = s2; conv_out[base + 3] = s3;
}

// ---------------------------------------------------------------------------
// LayerNorm over DM=256, one block per row; optional residual; in-place safe.
// Writes f32 (residual/head path) and bf16 (GEMM operand path).
// ---------------------------------------------------------------------------
__global__ __launch_bounds__(256)
void ln_kernel(const float* __restrict__ src, const float* __restrict__ res,
               float* __restrict__ dst, __bf16* __restrict__ dst_bf,
               const float* __restrict__ g, const float* __restrict__ beta) {
    __shared__ float red[256];
    const int m = blockIdx.x, t = threadIdx.x;
    float v = src[(size_t)m * DM + t];
    if (res) v += res[(size_t)m * DM + t];
    red[t] = v; __syncthreads();
    for (int s = 128; s > 0; s >>= 1) { if (t < s) red[t] += red[t + s]; __syncthreads(); }
    float mu = red[0] * (1.f / DM); __syncthreads();
    float d = v - mu;
    red[t] = d * d; __syncthreads();
    for (int s = 128; s > 0; s >>= 1) { if (t < s) red[t] += red[t + s]; __syncthreads(); }
    float var = red[0] * (1.f / DM);
    float o = d * rsqrtf(var + 1e-5f) * g[t] + beta[t];
    dst[(size_t)m * DM + t]    = o;
    dst_bf[(size_t)m * DM + t] = (__bf16)o;
}

// ---------------------------------------------------------------------------
// Causal depthwise conv (width 4, left pad 3) + bias + SiLU. u = xr[:, :DI].
// Writes f32 (scan path) and bf16 (x_proj GEMM path).
// ---------------------------------------------------------------------------
__global__ void dwconv_silu_kernel(const float* __restrict__ xr, const float* __restrict__ cw,
                                   const float* __restrict__ cb, float* __restrict__ u_act,
                                   __bf16* __restrict__ u_bf) {
    int idx = blockIdx.x * 256 + threadIdx.x;
    if (idx >= MM * DI) return;
    int d = idx % DI;
    int m = idx / DI;       // b*LL + l
    int l = m % LL;
    float s = cb[d];
    const float* __restrict__ w = cw + d * 4;
#pragma unroll
    for (int t = 0; t < 4; ++t) {
        int lt = l - 3 + t;
        if (lt >= 0) s += w[t] * xr[(size_t)(m - 3 + t) * (2 * DI) + d];
    }
    float o = silu_f(s);
    u_act[(size_t)m * DI + d] = o;
    u_bf[(size_t)m * DI + d]  = (__bf16)o;
}

// dt = softplus(dt_raw @ dtp_w^T + dtp_b), K=16 (below WMMA K: VALU).
__global__ void dtproj_kernel(const float* __restrict__ xdbl, const float* __restrict__ dtp_w,
                              const float* __restrict__ dtp_b, float* __restrict__ dt) {
    int idx = blockIdx.x * 256 + threadIdx.x;
    if (idx >= MM * DI) return;
    int n = idx % DI, m = idx / DI;
    const float* __restrict__ xrow = xdbl + (size_t)m * 48;
    const float* __restrict__ wrow = dtp_w + n * DTRK;
    float s = dtp_b[n];
#pragma unroll
    for (int k = 0; k < DTRK; ++k) s += xrow[k] * wrow[k];
    dt[idx] = softplus_f(s);
}

// ---------------------------------------------------------------------------
// Selective scan: one thread per (b, d); h[16] state in registers; 196 steps.
// 16384 threads = 512 waves -> serial-chain latency hidden; exp co-executes.
// ---------------------------------------------------------------------------
__global__ __launch_bounds__(256)
void scan_kernel(const float* __restrict__ u, const float* __restrict__ dt,
                 const float* __restrict__ xdbl, const float* __restrict__ A_log,
                 const float* __restrict__ Dp, float* __restrict__ y) {
    int tid = blockIdx.x * 256 + threadIdx.x;
    if (tid >= BB * DI) return;
    int d = tid % DI, b = tid / DI;
    float A[NS], h[NS];
#pragma unroll
    for (int n = 0; n < NS; ++n) { A[n] = -__expf(A_log[d * NS + n]); h[n] = 0.f; }
    const float Dd = Dp[d];
    for (int l = 0; l < LL; ++l) {
        size_t mb = (size_t)b * LL + l;
        float dtv = dt[mb * DI + d];
        float uv  = u[mb * DI + d];
        const float* __restrict__ Bm = xdbl + mb * 48 + 16;
        const float* __restrict__ Cm = xdbl + mb * 48 + 32;
        float dtu = dtv * uv;
        float yv = 0.f;
#pragma unroll
        for (int n = 0; n < NS; ++n) {
            h[n] = __expf(dtv * A[n]) * h[n] + dtu * Bm[n];
            yv += h[n] * Cm[n];
        }
        y[mb * DI + d] = yv + uv * Dd;
    }
}

// g = y * silu(z); g only feeds the out-proj GEMM -> bf16 only
__global__ void gate_kernel(const float* __restrict__ y, const float* __restrict__ xr,
                            __bf16* __restrict__ g) {
    int idx = blockIdx.x * 256 + threadIdx.x;
    if (idx >= MM * DI) return;
    int d = idx % DI, m = idx / DI;
    float z = xr[(size_t)m * (2 * DI) + DI + d];
    g[idx] = (__bf16)(y[idx] * silu_f(z));
}

// mean over L then 10-way head
__global__ __launch_bounds__(256)
void head_kernel(const float* __restrict__ h, const float* __restrict__ out_w,
                 const float* __restrict__ out_b, float* __restrict__ out) {
    __shared__ float pooled[DM];
    const int b = blockIdx.x, t = threadIdx.x;
    float s = 0.f;
    for (int l = 0; l < LL; ++l) s += h[((size_t)b * LL + l) * DM + t];
    pooled[t] = s * (1.f / LL);
    __syncthreads();
    if (t < DOUT) {
        float o = out_b[t];
        const float* __restrict__ wr = out_w + t * DM;
        for (int k = 0; k < DM; ++k) o += pooled[k] * wr[k];
        out[b * DOUT + t] = o;
    }
}

// ---------------------------------------------------------------------------
extern "C" void kernel_launch(void* const* d_in, const int* in_sizes, int n_in,
                              void* d_out, int out_size, void* d_ws, size_t ws_size,
                              hipStream_t stream) {
    const float* x     = (const float*)d_in[0];
    const float* pe_w  = (const float*)d_in[1];
    const float* pe_b  = (const float*)d_in[2];
    const float* pn_g  = (const float*)d_in[3];
    const float* pn_b  = (const float*)d_in[4];
    const float* in_w  = (const float*)d_in[5];
    const float* cw    = (const float*)d_in[6];
    const float* cb    = (const float*)d_in[7];
    const float* xp_w  = (const float*)d_in[8];
    const float* dtp_w = (const float*)d_in[9];
    const float* dtp_b = (const float*)d_in[10];
    const float* A_log = (const float*)d_in[11];
    const float* Dp    = (const float*)d_in[12];
    const float* op_w  = (const float*)d_in[13];
    const float* ln_g  = (const float*)d_in[14];
    const float* ln_b  = (const float*)d_in[15];
    const float* out_w = (const float*)d_in[16];
    const float* out_b = (const float*)d_in[17];

    // -------- workspace layout: f32 region then bf16 region (~98 MB) --------
    float* ws   = (float*)d_ws;
    float* h    = ws;                         // MM*DM
    float* xr   = h    + (size_t)MM * DM;     // MM*2*DI
    float* uact = xr   + (size_t)MM * 2 * DI; // MM*DI
    float* xdbl = uact + (size_t)MM * DI;     // MM*48
    float* dt   = xdbl + (size_t)MM * 48;     // MM*DI
    float* y    = dt   + (size_t)MM * DI;     // MM*DI
    float* tmp  = y    + (size_t)MM * DI;     // MM*DM

    __bf16* h_bf   = (__bf16*)(tmp + (size_t)MM * DM);
    __bf16* u_bf   = h_bf   + (size_t)MM * DM;          // MM*DI
    __bf16* g_bf   = u_bf   + (size_t)MM * DI;          // MM*DI
    __bf16* inw_bf = g_bf   + (size_t)MM * DI;          // NL*2*DI*DM
    __bf16* xpw_bf = inw_bf + (size_t)NL * 2 * DI * DM; // NL*48*DI
    __bf16* opw_bf = xpw_bf + (size_t)NL * 48 * DI;     // NL*DM*DI

    const int elems  = MM * DI;
    const int blocks = (elems + 255) / 256;

    // one-time weight conversion to bf16 (all layers)
    {
        int n1 = NL * 2 * DI * DM, n2 = NL * 48 * DI, n3 = NL * DM * DI;
        cvt_bf16_kernel<<<(n1 + 255) / 256, 256, 0, stream>>>(in_w, inw_bf, n1);
        cvt_bf16_kernel<<<(n2 + 255) / 256, 256, 0, stream>>>(xp_w, xpw_bf, n2);
        cvt_bf16_kernel<<<(n3 + 255) / 256, 256, 0, stream>>>(op_w, opw_bf, n3);
    }

    // patch embed (conv result stored channel-major == raw reshape) + LN
    patch_embed_kernel<<<BB * 49, 256, 0, stream>>>(x, pe_w, pe_b, tmp);
    ln_kernel<<<MM, 256, 0, stream>>>(tmp, nullptr, h, h_bf, pn_g, pn_b);

    for (int i = 0; i < NL; ++i) {
        // xr = h @ in_w^T  (6272 x 1024, K=256): 2x4 tile blocking
        gemm_bf16_wmma<2, 4><<<dim3(MM / 32, (2 * DI) / 64), 32, 0, stream>>>(
            h_bf, inw_bf + (size_t)i * 2 * DI * DM, xr, MM, 2 * DI, DM);
        // u = silu(causal_dwconv(xr[:, :DI]))
        dwconv_silu_kernel<<<blocks, 256, 0, stream>>>(
            xr, cw + (size_t)i * DI * 4, cb + (size_t)i * DI, uact, u_bf);
        // xdbl = u @ xp_w^T  (6272 x 48, K=512): 2x3 tile blocking
        gemm_bf16_wmma<2, 3><<<dim3(MM / 32, 1), 32, 0, stream>>>(
            u_bf, xpw_bf + (size_t)i * 48 * DI, xdbl, MM, 48, DI);
        // dt = softplus(xdbl[:, :16] @ dtp_w^T + dtp_b)
        dtproj_kernel<<<blocks, 256, 0, stream>>>(
            xdbl, dtp_w + (size_t)i * DI * DTRK, dtp_b + (size_t)i * DI, dt);
        // selective scan
        scan_kernel<<<(BB * DI + 255) / 256, 256, 0, stream>>>(
            uact, dt, xdbl, A_log + (size_t)i * DI * NS, Dp + (size_t)i * DI, y);
        // g = y * silu(z)
        gate_kernel<<<blocks, 256, 0, stream>>>(y, xr, g_bf);
        // tmp = g @ op_w^T  (6272 x 256, K=512): 2x4 tile blocking
        gemm_bf16_wmma<2, 4><<<dim3(MM / 32, DM / 64), 32, 0, stream>>>(
            g_bf, opw_bf + (size_t)i * DM * DI, tmp, MM, DM, DI);
        // h = LN(h + tmp)   (in-place safe: per-row, read-then-write)
        ln_kernel<<<MM, 256, 0, stream>>>(tmp, h, h, h_bf,
                                          ln_g + (size_t)i * DM, ln_b + (size_t)i * DM);
    }

    head_kernel<<<BB, 256, 0, stream>>>(h, out_w, out_b, (float*)d_out);
}
